// HAB_43181601194627
// MI455X (gfx1250) — matrix operand order repaired and checked by reading, not compiled, not run on hardware
//
#include <hip/hip_runtime.h>
#include <hip/hip_bf16.h>
#include <math.h>

typedef unsigned short u16;
typedef __attribute__((ext_vector_type(16))) __bf16 v16bf;
typedef __attribute__((ext_vector_type(8)))  float  v8f;

// ---- problem constants ----
constexpr int Bn = 2, Hd = 256, Wd = 256, Cn = 192;
constexpr int WS = 16, SH = 8, NHh = 6;      // head dim = 32
constexpr long NtokL = (long)Bn * Hd * Wd;   // 131072 rows
constexpr int QKVN = 3 * Cn;                 // 576
constexpr int MLPH = 4 * Cn;                 // 768

// ---- bf16 helpers (storage = u16, RNE rounding) ----
__device__ __forceinline__ u16 f2bf(float f) {
  unsigned u = __builtin_bit_cast(unsigned, f);
  u += 0x7FFFu + ((u >> 16) & 1u);
  return (u16)(u >> 16);
}
__device__ __forceinline__ float bf2f(u16 h) {
  unsigned u = ((unsigned)h) << 16;
  return __builtin_bit_cast(float, u);
}

// ---- WMMA fragment loaders (CDNA5 16x16x32 bf16 layouts) ----
// A (16x32): lane m=lane&15; two contiguous runs of 8 at K0+8*half and K0+16+8*half
__device__ __forceinline__ v16bf a_frag(const u16* rowptr /* &A[m][k0] */) {
  int half = (threadIdx.x >> 4) & 1;
  union { v16bf v; uint4 q[2]; } r;
  const u16* p = rowptr + half * 8;
  r.q[0] = *(const uint4*)(p);
  r.q[1] = *(const uint4*)(p + 16);
  return r.v;
}
// B (32x16): lane n=lane&15; two contiguous runs of 8 at K0+8*half and K0+16+8*half (B stored N-major: BT[n][k])
__device__ __forceinline__ v16bf b_frag(const u16* colptr /* &BT[n][k0] */) {
  int half = (threadIdx.x >> 4) & 1;
  union { v16bf v; uint4 q[2]; } r;
  const u16* p = colptr + half * 8;
  r.q[0] = *(const uint4*)(p);
  r.q[1] = *(const uint4*)(p + 16);
  return r.v;
}
__device__ __forceinline__ v8f wmma_bf(v16bf a, v16bf b, v8f c) {
  return __builtin_amdgcn_wmma_f32_16x16x32_bf16(false, a, false, b, (short)0, c, false, false);
}
__device__ __forceinline__ float gelu_exact(float v) {
  return 0.5f * v * (1.f + erff(v * 0.70710678118654752f));
}

// ---- Tensor Data Mover staging (gfx1250): DMA [rows x 192] bf16 strip -> LDS ----
#if __has_builtin(__builtin_amdgcn_tensor_load_to_lds) && __has_builtin(__builtin_amdgcn_s_wait_tensorcnt)
#define USE_TDM 1
typedef __attribute__((ext_vector_type(4))) unsigned int u32x4;
typedef __attribute__((ext_vector_type(8))) int          i32x8;
typedef __attribute__((ext_vector_type(4))) int          i32x4;

__device__ __forceinline__ void tdm_load_rows192(const u16* gsrc, u16* lds_dst, int rows) {
  unsigned long long ga = (unsigned long long)gsrc;
  unsigned ldsa = (unsigned)(unsigned long long)lds_dst;  // LDS byte offset (addr[31:0])
  u32x4 g0;
  g0[0] = 1u;                                   // count=1 (valid user descriptor)
  g0[1] = ldsa;                                 // lds_addr
  g0[2] = (unsigned)ga;                         // global_addr[31:0]
  g0[3] = ((unsigned)(ga >> 32) & 0x01FFFFFFu)  // global_addr[56:32]
        | (2u << 30);                           // type=2 ("image")
  i32x8 g1;
  g1[0] = (int)(1u << 16);                      // data_size=2B (code 1); no multicast/pad/iter
  g1[1] = (int)((192u & 0xFFFFu) << 16);        // tensor_dim0[15:0] @ bits 63:48
  g1[2] = (int)(((192u >> 16) & 0xFFFFu)        // tensor_dim0[31:16]
        | (((unsigned)131072 & 0xFFFFu) << 16));// tensor_dim1[15:0]
  g1[3] = (int)((((unsigned)131072 >> 16) & 0xFFFFu) // tensor_dim1[31:16]
        | (192u << 16));                        // tile_dim0 = 192
  g1[4] = rows;                                 // tile_dim1 = rows (tile_dim2 = 0)
  g1[5] = 192;                                  // tensor_dim0_stride[31:0]
  g1[6] = 0;                                    // stride0 hi | tensor_dim1_stride lo (unused 2D)
  g1[7] = 0;
  i32x4 z4 = {0, 0, 0, 0};                      // groups 2/3 unused (<=2D tile)
  i32x8 z8 = {0, 0, 0, 0, 0, 0, 0, 0};
  // 6-arg form (clang-23 / therock-10.0 headers): g0, g1, g2, g3, extra, cpol
  __builtin_amdgcn_tensor_load_to_lds(g0, g1, z4, z4, z8, 0);
}
#else
#define USE_TDM 0
#endif

// stage [16 x 192] bf16 A-strip to LDS: TDM from wave 0, else cooperative copy
__device__ __forceinline__ void stage_a16(const u16* gsrc, u16* lds_dst, int nthreads) {
#if USE_TDM
  if (threadIdx.x < 32) {
    tdm_load_rows192(gsrc, lds_dst, 16);
    __builtin_amdgcn_s_wait_tensorcnt(0);
  }
#else
  for (int i = threadIdx.x; i < 16 * Cn; i += nthreads) lds_dst[i] = gsrc[i];
#endif
  __syncthreads();
}

// ---- weight convert + transpose: src (K x N row-major f32) -> dst (N x K bf16) ----
__global__ void k_cvtT(const float* __restrict__ src, u16* __restrict__ dst, int K, int N) {
  long i = (long)blockIdx.x * 256 + threadIdx.x;
  if (i >= (long)K * N) return;
  int k = (int)(i / N), n = (int)(i % N);
  dst[(long)n * K + k] = f2bf(src[i]);
}

// ---- LayerNorm over C=192, 8 rows/block, 1 wave/row; bf16 out ----
__global__ void __launch_bounds__(256) k_ln(const float* __restrict__ x,
                                            const float* __restrict__ g,
                                            const float* __restrict__ bb,
                                            u16* __restrict__ out) {
  int row  = blockIdx.x * 8 + (threadIdx.x >> 5);
  int lane = threadIdx.x & 31;
  const float* xr = x + (long)row * Cn;
  float v[6]; float s = 0.f;
#pragma unroll
  for (int i = 0; i < 6; i++) { v[i] = xr[lane + i * 32]; s += v[i]; }
#pragma unroll
  for (int m = 16; m; m >>= 1) s += __shfl_xor(s, m);
  float mu = s * (1.f / 192.f);
  float var = 0.f;
#pragma unroll
  for (int i = 0; i < 6; i++) { float d = v[i] - mu; var += d * d; }
#pragma unroll
  for (int m = 16; m; m >>= 1) var += __shfl_xor(var, m);
  float rstd = rsqrtf(var * (1.f / 192.f) + 1e-5f);
#pragma unroll
  for (int i = 0; i < 6; i++) {
    int c = lane + i * 32;
    out[(long)row * Cn + c] = f2bf((v[i] - mu) * rstd * g[c] + bb[c]);
  }
}

// ---- QKV GEMM: (Ntok x 192) @ (192 x 576) + b -> bf16 (A via TDM->LDS) ----
__global__ void __launch_bounds__(128) k_gemm_qkv(const u16* __restrict__ A,
                                                  const u16* __restrict__ BT,
                                                  const float* __restrict__ bias,
                                                  u16* __restrict__ out) {
  __shared__ u16 As[16 * Cn];
  int wave = threadIdx.x >> 5, lane = threadIdx.x & 31;
  int m = lane & 15, half = lane >> 4;
  long m0 = (long)blockIdx.x * 16;
  int  n0 = blockIdx.y * 64 + wave * 16;
  stage_a16(A + m0 * Cn, As, 128);
  const u16* arow = As + m * Cn;
  const u16* bcol = BT + (long)(n0 + m) * Cn;
  v8f acc = {};
#pragma unroll
  for (int k0 = 0; k0 < Cn; k0 += 32)
    acc = wmma_bf(a_frag(arow + k0), b_frag(bcol + k0), acc);
  int nn = n0 + m;
#pragma unroll
  for (int r = 0; r < 8; r++) {
    long mm = m0 + r + 8 * half;
    out[mm * QKVN + nn] = f2bf(acc[r] + bias[nn]);
  }
}

// ---- conv1 implicit GEMM: 16 pixels x 64 out-ch, K = 9*192 = 1728, GELU ----
__global__ void __launch_bounds__(128) k_conv1(const u16* __restrict__ xn,
                                               const u16* __restrict__ w1T,
                                               const float* __restrict__ b1,
                                               u16* __restrict__ out) {
  __shared__ u16 patch[3 * 18 * Cn];  // 20.7KB
  int blk = blockIdx.x;               // 8192
  int b = blk >> 12, iy = (blk >> 4) & 255, x0 = (blk & 15) << 4;
  int tid = threadIdx.x;
  for (int i = tid; i < 3 * 18 * Cn; i += 128) {
    int ci = i % Cn; int t = i / Cn; int cc = t % 18; int r = t / 18;
    int yy = iy + r - 1, xx = x0 + cc - 1;
    u16 v = 0;
    if (yy >= 0 && yy < Hd && xx >= 0 && xx < Wd)
      v = xn[(((long)b * Hd + yy) * Wd + xx) * Cn + ci];
    patch[i] = v;
  }
  __syncthreads();
  int wave = tid >> 5, lane = tid & 31;
  int m = lane & 15, half = lane >> 4;
  int n0 = wave * 16;
  const u16* bcol = w1T + (long)(n0 + m) * 1728;
  v8f acc = {};
  for (int s = 0; s < 54; s++) {
    int kyx = s / 6, ci0 = (s % 6) * 32;
    int ky = kyx / 3, kx = kyx - ky * 3;
    acc = wmma_bf(a_frag(patch + (ky * 18 + m + kx) * Cn + ci0),
                  b_frag(bcol + kyx * Cn + ci0), acc);
  }
  long pix = ((long)b * Hd + iy) * Wd + x0;
  int nn = n0 + m;
#pragma unroll
  for (int r = 0; r < 8; r++) {
    float v = gelu_exact(acc[r] + b1[nn]);
    out[(pix + r + 8 * half) * 64 + nn] = f2bf(v);
  }
}

// ---- conv2 implicit GEMM: 16 pixels x 192 out-ch, K = 9*64 = 576 -> f32 cv ----
__global__ void __launch_bounds__(128) k_conv2(const u16* __restrict__ c1,
                                               const u16* __restrict__ w2T,
                                               const float* __restrict__ b2,
                                               float* __restrict__ cv) {
  __shared__ u16 patch[3 * 18 * 64];  // 6.9KB
  int blk = blockIdx.x;
  int b = blk >> 12, iy = (blk >> 4) & 255, x0 = (blk & 15) << 4;
  int tid = threadIdx.x;
  for (int i = tid; i < 3 * 18 * 64; i += 128) {
    int ci = i & 63; int t = i >> 6; int cc = t % 18; int r = t / 18;
    int yy = iy + r - 1, xx = x0 + cc - 1;
    u16 v = 0;
    if (yy >= 0 && yy < Hd && xx >= 0 && xx < Wd)
      v = c1[(((long)b * Hd + yy) * Wd + xx) * 64 + ci];
    patch[i] = v;
  }
  __syncthreads();
  int wave = tid >> 5, lane = tid & 31;
  int m = lane & 15, half = lane >> 4;
  long pix = ((long)b * Hd + iy) * Wd + x0;
  for (int nt = wave; nt < 12; nt += 4) {
    int n0 = nt * 16, nn = n0 + m;
    const u16* bcol = w2T + (long)nn * 576;
    v8f acc = {};
    for (int s = 0; s < 18; s++) {
      int kyx = s >> 1, ci0 = (s & 1) * 32;
      int ky = kyx / 3, kx = kyx - ky * 3;
      acc = wmma_bf(a_frag(patch + (ky * 18 + m + kx) * 64 + ci0),
                    b_frag(bcol + kyx * 64 + ci0), acc);
    }
#pragma unroll
    for (int r = 0; r < 8; r++)
      cv[(pix + r + 8 * half) * Cn + nn] = acc[r] + b2[nn];
  }
}

// ---- channel-attention helpers ----
__global__ void k_zero(float* p, int n) {
  int i = blockIdx.x * blockDim.x + threadIdx.x;
  if (i < n) p[i] = 0.f;
}
__global__ void __launch_bounds__(192) k_pool(const float* __restrict__ cv, float* pooled) {
  int b = blockIdx.x >> 8;
  long r0 = (long)(blockIdx.x & 255) * 256;
  int c = threadIdx.x;
  float s = 0.f;
  long base = ((long)b * 65536 + r0) * Cn + c;
  for (int i = 0; i < 256; i++) s += cv[base + (long)i * Cn];
  atomicAdd(&pooled[b * Cn + c], s);
}
__global__ void __launch_bounds__(192) k_ca(const float* __restrict__ pooled,
                                            const float* __restrict__ w1, const float* __restrict__ bb1,
                                            const float* __restrict__ w2, const float* __restrict__ bb2,
                                            float* __restrict__ yv) {
  int b = blockIdx.x, c = threadIdx.x;
  __shared__ float pm[Cn];
  __shared__ float t[NHh];
  pm[c] = pooled[b * Cn + c] * (1.f / 65536.f);
  __syncthreads();
  if (c < NHh) {
    float s = bb1[c];
    for (int i = 0; i < Cn; i++) s += pm[i] * w1[i * NHh + c];
    t[c] = fmaxf(s, 0.f);
  }
  __syncthreads();
  float s = bb2[c];
#pragma unroll
  for (int i = 0; i < NHh; i++) s += t[i] * w2[i * Cn + c];
  yv[b * Cn + c] = 1.f / (1.f + expf(-s));
}

// ---- shifted-window attention: one block per (window, head) ----
__global__ void __launch_bounds__(128) k_attn(const u16* __restrict__ qkv,
                                              const int* __restrict__ rpi,
                                              const float* __restrict__ mask,
                                              const float* __restrict__ rpb,
                                              u16* __restrict__ out) {
  __shared__ u16 Kl[256 * 32];      // K: [tok][d]     16KB
  __shared__ u16 Vt[32 * 256];      // V^T: [d][tok]   16KB
  __shared__ u16 Pl[4][16 * 256];   // per-wave P strip 32KB
  int wi = blockIdx.x / NHh;        // 0..511
  int h  = blockIdx.x % NHh;
  int tid = threadIdx.x, wave = tid >> 5, lane = tid & 31;
  int b = wi >> 8, wrow = (wi >> 4) & 15, wcol = wi & 15;

  // stage K and V^T for this window/head
  for (int t = tid; t < 256; t += 128) {
    int iy = wrow * 16 + (t >> 4), ix = wcol * 16 + (t & 15);
    long tok = ((long)b * Hd + ((iy + SH) & 255)) * Wd + ((ix + SH) & 255);
    const u16* kp = qkv + tok * QKVN + Cn + h * 32;
    const u16* vp = qkv + tok * QKVN + 2 * Cn + h * 32;
    uint4* kd = (uint4*)&Kl[t * 32];
    const uint4* ks = (const uint4*)kp;
    kd[0] = ks[0]; kd[1] = ks[1]; kd[2] = ks[2]; kd[3] = ks[3];
    for (int d = 0; d < 32; d++) Vt[d * 256 + t] = vp[d];
  }
  __syncthreads();

  int m = lane & 15, half = lane >> 4;
  const float qscale = 0.17677669529663687f;  // 32^-0.5
  u16* Pw = Pl[wave];

  for (int mt = wave; mt < 16; mt += 4) {
    // Q A-fragment straight from global (row = this lane's window token)
    int tq = mt * 16 + m;
    int iyq = wrow * 16 + (tq >> 4), ixq = wcol * 16 + (tq & 15);
    long tokq = ((long)b * Hd + ((iyq + SH) & 255)) * Wd + ((ixq + SH) & 255);
    v16bf aq = a_frag(qkv + tokq * QKVN + h * 32);

    float s[16][8];
    float rmax[8], rsum[8];
#pragma unroll
    for (int r = 0; r < 8; r++) rmax[r] = -1e30f;

    // scores: 16 tiles of 16x16, K=32 => one WMMA each; add RPB + mask
#pragma unroll
    for (int nt = 0; nt < 16; nt++) {
      v8f z = {};
      v8f c = wmma_bf(aq, b_frag(Kl + (nt * 16 + m) * 32), z);
      int nj = nt * 16 + m;
#pragma unroll
      for (int r = 0; r < 8; r++) {
        int mi = mt * 16 + r + 8 * half;
        float val = c[r] * qscale
                  + rpb[(long)rpi[mi * 256 + nj] * NHh + h]
                  + mask[((long)(wi & 255)) * 65536 + mi * 256 + nj];
        s[nt][r] = val;
        rmax[r] = fmaxf(rmax[r], val);
      }
    }
#pragma unroll
    for (int r = 0; r < 8; r++)
#pragma unroll
      for (int mk = 8; mk; mk >>= 1) rmax[r] = fmaxf(rmax[r], __shfl_xor(rmax[r], mk));
#pragma unroll
    for (int r = 0; r < 8; r++) rsum[r] = 0.f;
    // unnormalized exp -> per-wave LDS strip (normalize at output)
#pragma unroll
    for (int nt = 0; nt < 16; nt++) {
      int nj = nt * 16 + m;
#pragma unroll
      for (int r = 0; r < 8; r++) {
        float e = expf(s[nt][r] - rmax[r]);
        rsum[r] += e;
        Pw[(r + 8 * half) * 256 + nj] = f2bf(e);
      }
    }
#pragma unroll
    for (int r = 0; r < 8; r++)
#pragma unroll
      for (int mk = 8; mk; mk >>= 1) rsum[r] += __shfl_xor(rsum[r], mk);

    // P @ V : output 16x32 (two N tiles), K = 256 -> 8 WMMA steps each
    v8f o0 = {}, o1 = {};
#pragma unroll
    for (int ks = 0; ks < 8; ks++) {
      v16bf ap = a_frag(Pw + m * 256 + ks * 32);
      o0 = wmma_bf(ap, b_frag(Vt + (0 * 16 + m) * 256 + ks * 32), o0);
      o1 = wmma_bf(ap, b_frag(Vt + (1 * 16 + m) * 256 + ks * 32), o1);
    }
    // write back to un-shifted, un-windowed position (inverse roll fused)
#pragma unroll
    for (int r = 0; r < 8; r++) {
      int mi = mt * 16 + r + 8 * half;
      int iy2 = wrow * 16 + (mi >> 4), ix2 = wcol * 16 + (mi & 15);
      long tok2 = ((long)b * Hd + ((iy2 + SH) & 255)) * Wd + ((ix2 + SH) & 255);
      float inv = 1.f / rsum[r];
      out[tok2 * Cn + h * 32 + m]      = f2bf(o0[r] * inv);
      out[tok2 * Cn + h * 32 + 16 + m] = f2bf(o1[r] * inv);
    }
  }
}

// ---- proj GEMM + residual + scaled conv branch -> x2 (f32), A via TDM->LDS ----
__global__ void __launch_bounds__(128) k_proj(const u16* __restrict__ A,
                                              const u16* __restrict__ BT,
                                              const float* __restrict__ bias,
                                              const float* __restrict__ x_in,
                                              const float* __restrict__ cv,
                                              const float* __restrict__ yv,
                                              float* __restrict__ x2) {
  __shared__ u16 As[16 * Cn];
  int wave = threadIdx.x >> 5, lane = threadIdx.x & 31;
  int m = lane & 15, half = lane >> 4;
  long m0 = (long)blockIdx.x * 16;
  int  n0 = blockIdx.y * 64 + wave * 16;
  stage_a16(A + m0 * Cn, As, 128);
  const u16* arow = As + m * Cn;
  const u16* bcol = BT + (long)(n0 + m) * Cn;
  v8f acc = {};
#pragma unroll
  for (int k0 = 0; k0 < Cn; k0 += 32)
    acc = wmma_bf(a_frag(arow + k0), b_frag(bcol + k0), acc);
  int nn = n0 + m;
#pragma unroll
  for (int r = 0; r < 8; r++) {
    long mm = m0 + r + 8 * half;
    int bidx = (int)(mm >> 16);
    long o = mm * Cn + nn;
    x2[o] = x_in[o] + acc[r] + bias[nn] + cv[o] * yv[bidx * Cn + nn] * 0.01f;
  }
}

// ---- fused MLP: fc1 + GELU (to LDS bf16) + fc2 + residual -> d_out (f32) ----
__global__ void __launch_bounds__(256) k_mlp(const u16* __restrict__ hbf,
                                             const u16* __restrict__ fc1T,
                                             const float* __restrict__ bb1,
                                             const u16* __restrict__ fc2T,
                                             const float* __restrict__ bb2,
                                             const float* __restrict__ x2,
                                             float* __restrict__ out) {
  __shared__ u16 hm[16 * MLPH];  // 24KB
  __shared__ u16 As[16 * Cn];    // 6KB, TDM-staged fc1 input strip
  long m0 = (long)blockIdx.x * 16;
  int wave = threadIdx.x >> 5, lane = threadIdx.x & 31;
  int m = lane & 15, half = lane >> 4;
  stage_a16(hbf + m0 * Cn, As, 256);
  const u16* arow = As + m * Cn;
  for (int nt = wave; nt < 48; nt += 8) {
    int n0 = nt * 16, nn = n0 + m;
    const u16* bcol = fc1T + (long)nn * Cn;
    v8f acc = {};
#pragma unroll
    for (int k0 = 0; k0 < Cn; k0 += 32)
      acc = wmma_bf(a_frag(arow + k0), b_frag(bcol + k0), acc);
#pragma unroll
    for (int r = 0; r < 8; r++)
      hm[(r + 8 * half) * MLPH + nn] = f2bf(gelu_exact(acc[r] + bb1[nn]));
  }
  __syncthreads();
  for (int nt = wave; nt < 12; nt += 8) {
    int n0 = nt * 16, nn = n0 + m;
    const u16* bcol = fc2T + (long)nn * MLPH;
    v8f acc = {};
#pragma unroll
    for (int k0 = 0; k0 < MLPH; k0 += 32)
      acc = wmma_bf(a_frag(hm + m * MLPH + k0), b_frag(bcol + k0), acc);
#pragma unroll
    for (int r = 0; r < 8; r++) {
      long mm = m0 + r + 8 * half;
      out[mm * Cn + nn] = x2[mm * Cn + nn] + acc[r] + bb2[nn];
    }
  }
}

// =====================================================================
extern "C" void kernel_launch(void* const* d_in, const int* in_sizes, int n_in,
                              void* d_out, int out_size, void* d_ws, size_t ws_size,
                              hipStream_t stream) {
  const float* x        = (const float*)d_in[0];
  const int*   rpi      = (const int*)  d_in[1];
  const float* amask    = (const float*)d_in[2];
  const float* ln1_g    = (const float*)d_in[5];
  const float* ln1_b    = (const float*)d_in[6];
  const float* qkv_w    = (const float*)d_in[7];
  const float* qkv_b    = (const float*)d_in[8];
  const float* rpb      = (const float*)d_in[9];
  const float* proj_w   = (const float*)d_in[10];
  const float* proj_b   = (const float*)d_in[11];
  const float* conv1_w  = (const float*)d_in[12];
  const float* conv1_b  = (const float*)d_in[13];
  const float* conv2_w  = (const float*)d_in[14];
  const float* conv2_b  = (const float*)d_in[15];
  const float* ca1_w    = (const float*)d_in[16];
  const float* ca1_b    = (const float*)d_in[17];
  const float* ca2_w    = (const float*)d_in[18];
  const float* ca2_b    = (const float*)d_in[19];
  const float* ln2_g    = (const float*)d_in[20];
  const float* ln2_b    = (const float*)d_in[21];
  const float* fc1_w    = (const float*)d_in[22];
  const float* fc1_b    = (const float*)d_in[23];
  const float* fc2_w    = (const float*)d_in[24];
  const float* fc2_b    = (const float*)d_in[25];
  float* out = (float*)d_out;

  // workspace carve-up (256B aligned)
  char* ws = (char*)d_ws;
  size_t off = 0;
  auto take = [&](size_t bytes) { char* p = ws + off; off = (off + bytes + 255) & ~(size_t)255; return p; };
  u16*   qkv_bf  = (u16*)  take(NtokL * QKVN * 2);
  u16*   xn_bf   = (u16*)  take(NtokL * Cn * 2);
  u16*   attn_bf = (u16*)  take(NtokL * Cn * 2);
  u16*   h_bf    = (u16*)  take(NtokL * Cn * 2);
  float* x2      = (float*)take(NtokL * Cn * 4);
  float* cv      = (float*)take(NtokL * Cn * 4);
  u16*   c1      = (u16*)  take(NtokL * 64 * 2);
  u16*   qkv_wT  = (u16*)  take((size_t)Cn * QKVN * 2);
  u16*   proj_wT = (u16*)  take((size_t)Cn * Cn * 2);
  u16*   fc1_wT  = (u16*)  take((size_t)Cn * MLPH * 2);
  u16*   fc2_wT  = (u16*)  take((size_t)MLPH * Cn * 2);
  u16*   w1T     = (u16*)  take((size_t)1728 * 64 * 2);
  u16*   w2T     = (u16*)  take((size_t)576 * Cn * 2);
  float* pooled  = (float*)take(2 * Cn * 4);
  float* yv      = (float*)take(2 * Cn * 4);

  auto cvt = [&](const float* src, u16* dst, int K, int N) {
    long tot = (long)K * N;
    k_cvtT<<<dim3((unsigned)((tot + 255) / 256)), dim3(256), 0, stream>>>(src, dst, K, N);
  };
  // weights -> bf16, N-major
  cvt(qkv_w,  qkv_wT,  Cn,   QKVN);
  cvt(proj_w, proj_wT, Cn,   Cn);
  cvt(fc1_w,  fc1_wT,  Cn,   MLPH);
  cvt(fc2_w,  fc2_wT,  MLPH, Cn);
  cvt(conv1_w, w1T, 1728, 64);   // (ky,kx,ci) x co
  cvt(conv2_w, w2T, 576,  Cn);   // (ky,kx,ci) x co

  // LN1
  k_ln<<<dim3(16384), dim3(256), 0, stream>>>(x, ln1_g, ln1_b, xn_bf);
  // QKV
  k_gemm_qkv<<<dim3(8192, 9), dim3(128), 0, stream>>>(xn_bf, qkv_wT, qkv_b, qkv_bf);
  // CAB conv branch
  k_conv1<<<dim3(8192), dim3(128), 0, stream>>>(xn_bf, w1T, conv1_b, c1);
  k_conv2<<<dim3(8192), dim3(128), 0, stream>>>(c1, w2T, conv2_b, cv);
  k_zero<<<dim3(2), dim3(192), 0, stream>>>(pooled, 2 * Cn);
  k_pool<<<dim3(512), dim3(192), 0, stream>>>(cv, pooled);
  k_ca<<<dim3(2), dim3(192), 0, stream>>>(pooled, ca1_w, ca1_b, ca2_w, ca2_b, yv);
  // shifted-window attention (512 windows x 6 heads)
  k_attn<<<dim3(512 * NHh), dim3(128), 0, stream>>>(qkv_bf, rpi, amask, rpb, attn_bf);
  // proj + residual + conv-branch merge -> x2
  k_proj<<<dim3(8192, 3), dim3(128), 0, stream>>>(attn_bf, proj_wT, proj_b, x, cv, yv, x2);
  // LN2 + fused MLP -> out
  k_ln<<<dim3(16384), dim3(256), 0, stream>>>(x2, ln2_g, ln2_b, h_bf);
  k_mlp<<<dim3(8192), dim3(256), 0, stream>>>(h_bf, fc1_wT, fc1_b, fc2_wT, fc2_b, x2, out);
}
